// Inception3_DS_Hash_29609504539343
// MI455X (gfx1250) — compile-verified, matrix operand order checked by reading, not run on hardware
//
#include <hip/hip_runtime.h>
#include <stdint.h>
#include <stddef.h>

// ---------------- problem constants ----------------
constexpr int B_    = 32;
constexpr int CIN   = 256;
constexpr int COUT  = 512;
constexpr int HH    = 48;
constexpr int HW    = HH * HH;          // 2304
constexpr int DD    = 9 * CIN;          // 2304 (K*K*C_IN)
constexpr int TABLES = 10;
constexpr int HASHB  = 8;
constexpr int NCOLS  = B_ * HW;         // 73728

// ---------------- WMMA types ----------------
typedef __attribute__((ext_vector_type(16))) __bf16 v16bf;
typedef __attribute__((ext_vector_type(8)))  float  v8f;

// gfx1250 async global->LDS path (guarded: falls back to reg-staged copies)
#if defined(__gfx1250__) && __has_builtin(__builtin_amdgcn_global_load_async_to_lds_b128)
#define USE_ASYNC_LDS 1
typedef int v4i_raw __attribute__((vector_size(16)));            // matches builtin prototype
typedef __attribute__((address_space(1))) v4i_raw as1_v4i;       // global (prints as __device__)
typedef __attribute__((address_space(3))) v4i_raw as3_v4i;       // LDS
#else
#define USE_ASYNC_LDS 0
#endif

__device__ inline void wait_async_zero() {
#if USE_ASYNC_LDS
#if __has_builtin(__builtin_amdgcn_s_wait_asynccnt)
    __builtin_amdgcn_s_wait_asynccnt(0);
#else
    asm volatile("s_wait_asynccnt 0x0" ::: "memory");
#endif
#endif
}

__device__ inline __bf16 f2bf(float f) {
    union { float f; unsigned u; } in; in.f = f;
    unsigned r = in.u + 0x7FFFu + ((in.u >> 16) & 1u);  // round-to-nearest-even
    union { unsigned short s; __bf16 b; } out;
    out.s = (unsigned short)(r >> 16);
    return out.b;
}

// ---------------- stage 1: avg-pool query (batch element 0) ----------------
__global__ void pool_query_kernel(const float* __restrict__ x, float* __restrict__ q) {
    int out = blockIdx.x * blockDim.x + threadIdx.x;   // [0, 2304)
    if (out >= DD) return;
    int c = out / 9, j = out % 9, ph = j / 3, pw = j % 3;
    const float* base = x + ((size_t)c * HH + ph * 16) * HH + pw * 16;
    float s = 0.f;
    #pragma unroll 4
    for (int i = 0; i < 16; ++i)
        for (int jx = 0; jx < 16; ++jx)
            s += base[i * HH + jx];
    q[out] = s * (1.0f / 256.0f);
}

// ---------------- stage 2a: hash query ----------------
__global__ void hash_q_kernel(const float* __restrict__ rm, const float* __restrict__ q,
                              unsigned* __restrict__ sig_q) {
    __shared__ unsigned char bits[128];
    int i = threadIdx.x;                 // i = t*8 + h
    float s = 0.f;
    if (i < TABLES * HASHB) {
        const float* rv = rm + (size_t)i * DD;
        for (int d = 0; d < DD; ++d) s = fmaf(rv[d], q[d], s);
    }
    bits[i] = (s > 0.f) ? 1 : 0;
    __syncthreads();
    if ((i & 7) == 0 && i < TABLES * HASHB) {
        unsigned sig = 0;
        #pragma unroll
        for (int hh = 0; hh < 8; ++hh)
            sig |= (unsigned)bits[i + hh] << (7 - hh);   // MSB-first pack
        sig_q[i >> 3] = sig;
    }
}

// ---------------- stage 2b: hash all weight rows ----------------
__global__ __launch_bounds__(256) void hash_w_kernel(const float* __restrict__ rm,
                                                     const float* __restrict__ w,
                                                     unsigned* __restrict__ sig_w) {
    __shared__ unsigned char bits[256];
    int i = blockIdx.x * 256 + threadIdx.x;  // [0, 40960); i = (t*512+o)*8+h
    int t = i >> 12;
    int o = (i >> 3) & (COUT - 1);
    const float* rv = rm + ((size_t)t * HASHB + (i & 7)) * DD;
    const float* wv = w + (size_t)o * DD;
    float s = 0.f;
    for (int d = 0; d < DD; ++d) s = fmaf(rv[d], wv[d], s);
    bits[threadIdx.x] = (s > 0.f) ? 1 : 0;
    __syncthreads();
    if ((threadIdx.x & 7) == 0) {
        unsigned sig = 0;
        #pragma unroll
        for (int hh = 0; hh < 8; ++hh)
            sig |= (unsigned)bits[threadIdx.x + hh] << (7 - hh);
        sig_w[t * COUT + o] = sig;
    }
}

// ---------------- stage 3: votes + top-k(256) with JAX tie-break ----------------
__global__ __launch_bounds__(512) void vote_mask_kernel(const unsigned* __restrict__ sig_w,
                                                        const unsigned* __restrict__ sig_q,
                                                        float* __restrict__ mask) {
    __shared__ int votes[COUT];
    int o = threadIdx.x;
    int v = 0;
    #pragma unroll
    for (int t = 0; t < TABLES; ++t) v += (sig_w[t * COUT + o] == sig_q[t]) ? 1 : 0;
    votes[o] = v;
    __syncthreads();
    int r = 0;
    for (int o2 = 0; o2 < COUT; ++o2) {
        int v2 = votes[o2];
        r += ((v2 > v) || (v2 == v && o2 < o)) ? 1 : 0;
    }
    mask[o] = (v > 0 && r < (COUT / 2)) ? 1.f : 0.f;
}

// ---------------- stage 3b: pre-bake masked weights -> bf16, layout [tap][m][c] ----------------
__global__ __launch_bounds__(256) void prep_w_kernel(const float* __restrict__ w,
                                                     const float* __restrict__ mask,
                                                     __bf16* __restrict__ wbf) {
    int idx = blockIdx.x * 256 + threadIdx.x;     // [0, 9*512*256)
    int c   = idx & (CIN - 1);
    int m   = (idx >> 8) & (COUT - 1);
    int tap = idx >> 17;
    float v = w[((size_t)m * CIN + c) * 9 + tap] * mask[m];
    wbf[idx] = f2bf(v);
}

// ---------------- stage 4: masked 3x3 conv as implicit GEMM on bf16 WMMA ----------------
constexpr int MT = 128, NT = 128, KT = 32;
constexpr int LDSTR  = 40;               // padded LDS row stride in bf16 (80 B, 16B aligned)
constexpr int STAGES = 9 * (CIN / KT);   // 72

__global__ __launch_bounds__(256) void conv_wmma_kernel(const float* __restrict__ x,
                                                        const __bf16* __restrict__ wbf,
                                                        float* __restrict__ y) {
    __shared__ __align__(16) __bf16 aT[2][MT * LDSTR];   // A: [m][k], double buffered
    __shared__ __align__(16) __bf16 bT[2][NT * LDSTR];   // B: [n][k], double buffered

    const int tid  = threadIdx.x;
    const int lane = tid & 31;
    const int wv   = tid >> 5;
    const int wm   = wv & 3;        // 4 waves along M
    const int wn   = wv >> 2;       // 2 waves along N
    const int m0   = blockIdx.y * MT;
    const int ntile = blockIdx.x;
    const int b    = ntile / (HW / NT);
    const int p0   = (ntile % (HW / NT)) * NT;
    const float* xb = x + (size_t)b * CIN * HW;

    // stage-invariant per-thread load geometry
    const int n   = tid & 127;      // B column this thread owns (all 16 elements)
    const int kb  = tid >> 7;       // k parity: loads k = 2i + kb
    const int oh  = (p0 + n) / HH;
    const int ow  = (p0 + n) % HH;
    const int arow = tid >> 2;      // A rows arow, arow+64; 16B segment aseg
    const int aseg = tid & 3;

    float breg[16];
#if !USE_ASYNC_LDS
    int4 areg[2];
#endif

    auto issueA = [&](int tap, int c0, int buf) {
        #pragma unroll
        for (int j = 0; j < 2; ++j) {
            int row = arow + 64 * j;
            const __bf16* gp = wbf + ((size_t)tap * COUT + m0 + row) * CIN + c0 + aseg * 8;
            __bf16* lp = &aT[buf][row * LDSTR + aseg * 8];
#if USE_ASYNC_LDS
            __builtin_amdgcn_global_load_async_to_lds_b128(
                (as1_v4i*)gp, (as3_v4i*)lp, 0, 0);
#else
            areg[j] = *(const int4*)gp;
            (void)lp;
#endif
        }
    };
#if !USE_ASYNC_LDS
    auto storeA = [&](int buf) {
        #pragma unroll
        for (int j = 0; j < 2; ++j) {
            int row = arow + 64 * j;
            *(int4*)&aT[buf][row * LDSTR + aseg * 8] = areg[j];
        }
    };
#endif
    auto prefetchB = [&](int tap, int c0) {
        int dy = tap / 3 - 1, dx = tap % 3 - 1;
        int ih = oh + dy, iw = ow + dx;
        bool ok = (unsigned)ih < (unsigned)HH && (unsigned)iw < (unsigned)HH;
        const float* bp = xb + (size_t)(c0 + kb) * HW + ih * HH + iw;
        #pragma unroll
        for (int i = 0; i < 16; ++i)
            breg[i] = ok ? bp[(size_t)(2 * i) * HW] : 0.f;
    };
    auto storeB = [&](int buf) {
        #pragma unroll
        for (int i = 0; i < 16; ++i)
            bT[buf][n * LDSTR + 2 * i + kb] = f2bf(breg[i]);
    };

    v8f acc[2][4];
    #pragma unroll
    for (int mf = 0; mf < 2; ++mf)
        #pragma unroll
        for (int nf = 0; nf < 4; ++nf) {
            v8f z = {0.f,0.f,0.f,0.f,0.f,0.f,0.f,0.f};
            acc[mf][nf] = z;
        }

    const int half = lane >> 4;
    const int lrow = lane & 15;

    // ---- prologue: stage 0 into buffer 0 ----
    issueA(0, 0, 0);
    prefetchB(0, 0);
    storeB(0);
#if !USE_ASYNC_LDS
    storeA(0);
#endif

    for (int s = 0; s < STAGES; ++s) {
        const int buf = s & 1;
        wait_async_zero();            // stage-s async A landed in LDS
        __syncthreads();              // tiles for stage s visible; buf^1 free

        const bool more = (s + 1) < STAGES;
        if (more) {                   // overlap next stage's global traffic with WMMAs
            int nt = (s + 1) >> 3;
            int nc0 = ((s + 1) & 7) * KT;
            issueA(nt, nc0, buf ^ 1);
            prefetchB(nt, nc0);
        }

        // ---- fragment fetch (ds_load_b128 pairs) + 8 WMMAs ----
        v16bf afr[2], bfr[4];
        #pragma unroll
        for (int mf = 0; mf < 2; ++mf) {
            const int4* pr = (const int4*)(&aT[buf][(wm * 32 + mf * 16 + lrow) * LDSTR]);
            union { int4 qd[2]; v16bf v; } u;
            u.qd[0] = pr[half];
            u.qd[1] = pr[2 + half];
            afr[mf] = u.v;
        }
        #pragma unroll
        for (int nf = 0; nf < 4; ++nf) {
            const int4* pr = (const int4*)(&bT[buf][(wn * 64 + nf * 16 + lrow) * LDSTR]);
            union { int4 qd[2]; v16bf v; } u;
            u.qd[0] = pr[half];
            u.qd[1] = pr[2 + half];
            bfr[nf] = u.v;
        }
        #pragma unroll
        for (int mf = 0; mf < 2; ++mf)
            #pragma unroll
            for (int nf = 0; nf < 4; ++nf)
                acc[mf][nf] = __builtin_amdgcn_wmma_f32_16x16x32_bf16(
                    false, afr[mf], false, bfr[nf],
                    (short)0, acc[mf][nf], false, false);

        if (more) {                   // commit staged B (and A fallback) to the free buffer
            storeB(buf ^ 1);
#if !USE_ASYNC_LDS
            storeA(buf ^ 1);
#endif
        }
    }

    // ---- write out: VGPR r: lanes0-15 M=r, lanes16-31 M=r+8; N=lane&15 ----
    #pragma unroll
    for (int mf = 0; mf < 2; ++mf)
        #pragma unroll
        for (int nf = 0; nf < 4; ++nf) {
            int nn = p0 + wn * 64 + nf * 16 + lrow;
            #pragma unroll
            for (int r = 0; r < 8; ++r) {
                int m = m0 + wm * 32 + mf * 16 + r + 8 * half;
                y[((size_t)b * COUT + m) * HW + nn] = acc[mf][nf][r];
            }
        }
}

// ---------------- stage 5a: per-channel batch stats ----------------
__global__ __launch_bounds__(256) void bn_stats_kernel(const float* __restrict__ y,
                                                       float* __restrict__ meanOut,
                                                       float* __restrict__ rstdOut) {
    const int m = blockIdx.x;
    float s = 0.f, sq = 0.f;
    for (int i = threadIdx.x; i < B_ * HW; i += 256) {
        int b = i / HW, p = i - b * HW;
        float v = y[((size_t)b * COUT + m) * HW + p];
        s += v; sq += v * v;
    }
    __shared__ float ls[256], lq[256];
    ls[threadIdx.x] = s; lq[threadIdx.x] = sq;
    __syncthreads();
    for (int off = 128; off > 0; off >>= 1) {
        if (threadIdx.x < off) {
            ls[threadIdx.x] += ls[threadIdx.x + off];
            lq[threadIdx.x] += lq[threadIdx.x + off];
        }
        __syncthreads();
    }
    if (threadIdx.x == 0) {
        const float nT = (float)(B_ * HW);
        float mean = ls[0] / nT;
        float var  = lq[0] / nT - mean * mean;   // biased variance
        meanOut[m] = mean;
        rstdOut[m] = rsqrtf(var + 1e-3f);
    }
}

// ---------------- stage 5b: normalize + ReLU, vectorized in place ----------------
__global__ void bn_apply_kernel(float* __restrict__ y,
                                const float* __restrict__ gamma, const float* __restrict__ beta,
                                const float* __restrict__ mean, const float* __restrict__ rstd) {
    size_t total = (size_t)B_ * COUT * HW / 4;
    for (size_t i = (size_t)blockIdx.x * blockDim.x + threadIdx.x; i < total;
         i += (size_t)gridDim.x * blockDim.x) {
        float4 v = ((float4*)y)[i];
        int m = (int)((i * 4 / HW) % COUT);     // 2304 % 4 == 0 -> all 4 same channel
        float mu = mean[m], rs = rstd[m], gm = gamma[m], be = beta[m];
        v.x = fmaxf((v.x - mu) * rs * gm + be, 0.f);
        v.y = fmaxf((v.y - mu) * rs * gm + be, 0.f);
        v.z = fmaxf((v.z - mu) * rs * gm + be, 0.f);
        v.w = fmaxf((v.w - mu) * rs * gm + be, 0.f);
        ((float4*)y)[i] = v;
    }
}

// ---------------- launch ----------------
extern "C" void kernel_launch(void* const* d_in, const int* in_sizes, int n_in,
                              void* d_out, int out_size, void* d_ws, size_t ws_size,
                              hipStream_t stream) {
    const float* x       = (const float*)d_in[0];
    const float* whole_w = (const float*)d_in[1];
    const float* gamma   = (const float*)d_in[2];
    const float* beta    = (const float*)d_in[3];
    const float* rm      = (const float*)d_in[4];
    float* y = (float*)d_out;

    uint8_t* wsb = (uint8_t*)d_ws;
    float*    q     = (float*)   (wsb + 0);       // 2304 f32
    unsigned* sig_q = (unsigned*)(wsb + 9216);    // 10 u32
    unsigned* sig_w = (unsigned*)(wsb + 9472);    // 5120 u32
    float*    mask  = (float*)   (wsb + 29952);   // 512 f32
    float*    mean  = (float*)   (wsb + 32000);   // 512 f32
    float*    rstd  = (float*)   (wsb + 34048);   // 512 f32
    __bf16*   wbf   = (__bf16*)  (wsb + 36352);   // 9*512*256 bf16 (2.25 MB)

    pool_query_kernel<<<9, 256, 0, stream>>>(x, q);
    hash_q_kernel<<<1, 128, 0, stream>>>(rm, q, sig_q);
    hash_w_kernel<<<160, 256, 0, stream>>>(rm, whole_w, sig_w);
    vote_mask_kernel<<<1, 512, 0, stream>>>(sig_w, sig_q, mask);
    prep_w_kernel<<<(9 * COUT * CIN) / 256, 256, 0, stream>>>(whole_w, mask, wbf);
    conv_wmma_kernel<<<dim3(NCOLS / NT, COUT / MT), 256, 0, stream>>>(x, wbf, y);
    bn_stats_kernel<<<COUT, 256, 0, stream>>>(y, mean, rstd);
    bn_apply_kernel<<<2048, 256, 0, stream>>>(y, gamma, beta, mean, rstd);
}
